// Attention_28604482191653
// MI455X (gfx1250) — matrix operand order repaired and checked by reading, not compiled, not run on hardware
//
#include <hip/hip_runtime.h>

// ---------------------------------------------------------------------------
// MI455X (gfx1250) cosine-attention block.
// Compute-bound (~275 GFLOP vs ~0.5 GB HBM traffic @ 23.3 TB/s) => everything
// runs on v_wmma_f32_16x16x32_bf16 (wave32 WMMA, f32 accumulate).
// GEMM: 128x256 block tile, BK=32, 8 waves as 2(M)x4(N), 4x4 16x16 fragments
// per wave (16 WMMA / K-step), double-buffered LDS, async global->LDS DMA
// (ASYNCcnt-tracked, gfx1250) for the transposed-B GEMMs.
// ---------------------------------------------------------------------------

typedef __bf16 bf16;
typedef bf16  bf16x16 __attribute__((ext_vector_type(16)));
typedef float f32x8   __attribute__((ext_vector_type(8)));
typedef unsigned int u32x4 __attribute__((ext_vector_type(4)));
typedef int          s32x4 __attribute__((ext_vector_type(4)));

#define GLOBAL_AS __attribute__((address_space(1)))
#define LDS_AS    __attribute__((address_space(3)))

#if defined(__has_builtin)
#if __has_builtin(__builtin_amdgcn_global_load_async_to_lds_b128) && \
    __has_builtin(__builtin_amdgcn_s_wait_asynccnt)
#define USE_ASYNC_LDS 1
#endif
#endif
#ifndef USE_ASYNC_LDS
#define USE_ASYNC_LDS 0
#endif

struct U128x2  { u32x4 lo, hi; };   // 32 bytes
struct BFARR16 { bf16 h[16]; };     // 32 bytes

__device__ inline bf16x16 pack16(u32x4 lo, u32x4 hi) {
    U128x2 t{lo, hi};
    return __builtin_bit_cast(bf16x16, t);
}

// ---------------------------------------------------------------------------
// fp32 -> bf16 conversion (grid-stride)
// ---------------------------------------------------------------------------
__global__ void __launch_bounds__(256)
cvt_f32_bf16(const float* __restrict__ in, bf16* __restrict__ out, int n) {
    int i = blockIdx.x * blockDim.x + threadIdx.x;
    int stride = gridDim.x * blockDim.x;
    for (; i < n; i += stride) out[i] = (bf16)in[i];
}

// ---------------------------------------------------------------------------
// Row-wise L2 normalize (F.normalize, p=2, eps=1e-12), in place, bf16 rows.
// ---------------------------------------------------------------------------
__global__ void __launch_bounds__(256)
l2norm_rows(bf16* __restrict__ q, int d) {
    __shared__ float sb[256];
    bf16* row = q + (size_t)blockIdx.x * d;
    const int tid = threadIdx.x;
    float s = 0.0f;
    for (int c = tid; c < d; c += 256) { float v = (float)row[c]; s += v * v; }
    sb[tid] = s; __syncthreads();
    for (int st = 128; st > 0; st >>= 1) {
        if (tid < st) sb[tid] += sb[tid + st];
        __syncthreads();
    }
    const float scale = 1.0f / fmaxf(sqrtf(sb[0]), 1e-12f);
    for (int c = tid; c < d; c += 256) row[c] = (bf16)((float)row[c] * scale);
}

// ---------------------------------------------------------------------------
// Row-wise softmax: f32 in, bf16 out. One block (256 thr) per row of length n.
// ---------------------------------------------------------------------------
__global__ void __launch_bounds__(256)
softmax_rows(const float* __restrict__ sim, bf16* __restrict__ P, int n) {
    __shared__ float sb[256];
    const float* row = sim + (size_t)blockIdx.x * n;
    bf16* orow = P + (size_t)blockIdx.x * n;
    const int tid = threadIdx.x;
    float m = -3.4e38f;
    for (int c = tid; c < n; c += 256) m = fmaxf(m, row[c]);
    sb[tid] = m; __syncthreads();
    for (int st = 128; st > 0; st >>= 1) {
        if (tid < st) sb[tid] = fmaxf(sb[tid], sb[tid + st]);
        __syncthreads();
    }
    m = sb[0]; __syncthreads();
    float s = 0.0f;
    for (int c = tid; c < n; c += 256) s += __expf(row[c] - m);
    sb[tid] = s; __syncthreads();
    for (int st = 128; st > 0; st >>= 1) {
        if (tid < st) sb[tid] += sb[tid + st];
        __syncthreads();
    }
    const float inv = 1.0f / sb[0];
    for (int c = tid; c < n; c += 256) orow[c] = (bf16)(__expf(row[c] - m) * inv);
}

// ---------------------------------------------------------------------------
// WMMA GEMM:  C = act( A @ op(B) + bias ),  A: MxK bf16 row-major.
//   TRANS_B=1: B is NxK row-major (weight, use B^T)   -> async DMA path
//   TRANS_B=0: B is KxN row-major (V in attn@V)       -> reg/scatter path
// Block tile BM=128 x BN=256, BK=32. 256 threads = 8 wave32s, 2(M) x 4(N);
// each wave owns 4x4 fragments of 16x16 -> 16 v_wmma per K-step.
// Double-buffered LDS, one s_barrier per K-step. Batched via blockIdx.z.
// ---------------------------------------------------------------------------
template<bool TRANS_B, int ACT, typename CT>
__global__ void __launch_bounds__(256)
gemm_wmma(const bf16* __restrict__ A, const bf16* __restrict__ Bm,
          const float* __restrict__ bias, CT* __restrict__ C,
          int M, int N, int K, int lda, int ldb, int ldc,
          size_t sA, size_t sB, size_t sC)
{
    (void)M; (void)N;
    A  += (size_t)blockIdx.z * sA;
    Bm += (size_t)blockIdx.z * sB;
    C  += (size_t)blockIdx.z * sC;

    constexpr int BM = 128, BN = 256, BK = 32, LDP = BK + 8; // 40 bf16 = 80B pitch
    __shared__ bf16 Al[2][BM * LDP];
    __shared__ bf16 Bl[2][BN * LDP];

    const int tid  = threadIdx.x;
    const int lane = tid & 31;
    const int w    = tid >> 5;   // wave 0..7
    const int wy   = w >> 2;     // M direction: 2 waves * 64 rows
    const int wx   = w & 3;      // N direction: 4 waves * 64 cols
    const int blockM = blockIdx.y * BM;
    const int blockN = blockIdx.x * BN;

    // A (16x32) lane layout (matches B): lane<16 -> row=lane, K 0..15;
    // lane>=16 -> row=lane-16, K 16..31.
    const int rA    = (lane < 16) ? lane : (lane - 16);
    const int kOffA = (lane < 16) ? 0 : 16;
    // B (32x16) lane layout used here (same k order as A): lane n -> column n, K 0..15;
    // lane n+16 -> column n, K 16..31. (Bl stored [n][k], k-contiguous.)
    const int nB    = lane & 15;
    const int kOffB = (lane < 16) ? 0 : 16;

    f32x8 acc[4][4];
    #pragma unroll
    for (int i = 0; i < 4; ++i)
        #pragma unroll
        for (int j = 0; j < 4; ++j) acc[i][j] = f32x8{};

    const int aRow = tid >> 1;            // A tile: 2 threads per row
    const int aCol = (tid & 1) * 16;

    u32x4 ra[2], rb[4];
    auto loadRegs = [&](int k0) {
        const bf16* sAp = A + (size_t)(blockM + aRow) * lda + (k0 + aCol);
        ra[0] = *(const u32x4*)(sAp);
        ra[1] = *(const u32x4*)(sAp + 8);
        __builtin_prefetch(sAp + BK, 0, 1);       // global_prefetch_b8
        if constexpr (TRANS_B) {
            const bf16* sBp = Bm + (size_t)(blockN + tid) * ldb + k0; // one row/thread
            #pragma unroll
            for (int c = 0; c < 4; ++c) rb[c] = *(const u32x4*)(sBp + 8 * c);
            __builtin_prefetch(sBp + BK, 0, 1);
        } else {
            const bf16* sBp = Bm + (size_t)(k0 + (tid >> 3)) * ldb
                                 + (blockN + (tid & 7) * 32);
            #pragma unroll
            for (int c = 0; c < 4; ++c) rb[c] = *(const u32x4*)(sBp + 8 * c);
        }
    };
    auto storeLds = [&](int buf) {
        *(u32x4*)&Al[buf][aRow * LDP + aCol]     = ra[0];
        *(u32x4*)&Al[buf][aRow * LDP + aCol + 8] = ra[1];
        if constexpr (TRANS_B) {
            #pragma unroll
            for (int c = 0; c < 4; ++c)
                *(u32x4*)&Bl[buf][tid * LDP + 8 * c] = rb[c];
        } else {
            const int kB = tid >> 3;              // 0..31
            const int ncB = (tid & 7) * 32;       // 0..224
            #pragma unroll
            for (int cp = 0; cp < 2; ++cp) {      // transpose-scatter 32 cols
                U128x2 u{rb[2 * cp], rb[2 * cp + 1]};
                BFARR16 arr = __builtin_bit_cast(BFARR16, u);
                #pragma unroll
                for (int i = 0; i < 16; ++i)
                    Bl[buf][(ncB + 16 * cp + i) * LDP + kB] = arr.h[i];
            }
        }
    };

    auto computeTile = [&](int buf) {
        bf16x16 bfrag[4];
        #pragma unroll
        for (int j = 0; j < 4; ++j) {
            const bf16* p = &Bl[buf][(wx * 64 + j * 16 + nB) * LDP + kOffB];
            bfrag[j] = pack16(*(const u32x4*)p, *(const u32x4*)(p + 8));
        }
        #pragma unroll
        for (int i = 0; i < 4; ++i) {
            const bf16* p = &Al[buf][(wy * 64 + i * 16 + rA) * LDP + kOffA];
            bf16x16 afrag = pack16(*(const u32x4*)p, *(const u32x4*)(p + 8));
            #pragma unroll
            for (int j = 0; j < 4; ++j)
                acc[i][j] = __builtin_amdgcn_wmma_f32_16x16x32_bf16(
                    false, afrag, false, bfrag[j], (short)0, acc[i][j],
                    false, false);
        }
    };

#if USE_ASYNC_LDS
    // gfx1250 async global->LDS DMA path (ASYNCcnt-tracked), TRANS_B only.
    // Builtin prototype (from clang diagnostic): (v4i AS1*, v4i AS3*, imm, imm)
    auto asyncStage = [&](int buf, int k0) {
        const bf16* sAp = A + (size_t)(blockM + aRow) * lda + (k0 + aCol);
        bf16* dAp = &Al[buf][aRow * LDP + aCol];
        __builtin_amdgcn_global_load_async_to_lds_b128(
            (GLOBAL_AS s32x4*)sAp, (LDS_AS s32x4*)dAp, 0, 0);
        __builtin_amdgcn_global_load_async_to_lds_b128(
            (GLOBAL_AS s32x4*)(sAp + 8), (LDS_AS s32x4*)(dAp + 8), 0, 0);
        const bf16* sBp = Bm + (size_t)(blockN + tid) * ldb + k0;
        bf16* dBp = &Bl[buf][tid * LDP];
        #pragma unroll
        for (int c = 0; c < 4; ++c)
            __builtin_amdgcn_global_load_async_to_lds_b128(
                (GLOBAL_AS s32x4*)(sBp + 8 * c),
                (LDS_AS s32x4*)(dBp + 8 * c), 0, 0);
    };
    if constexpr (TRANS_B) {
        asyncStage(0, 0);
        int buf = 0;
        for (int k0 = 0; k0 < K; k0 += BK) {
            __builtin_amdgcn_s_wait_asynccnt(0);   // my DMA into `buf` landed
            __syncthreads();                       // everyone's landed
            if (k0 + BK < K) asyncStage(buf ^ 1, k0 + BK); // overlap with WMMA
            computeTile(buf);
            buf ^= 1;
        }
    } else
#endif
    {
        // register-staged double-buffered pipeline
        loadRegs(0);
        storeLds(0);
        int buf = 0;
        for (int k0 = 0; k0 < K; k0 += BK) {
            __syncthreads();
            const bool more = (k0 + BK) < K;
            if (more) loadRegs(k0 + BK);   // global loads issued before WMMAs
            computeTile(buf);
            if (more) storeLds(buf ^ 1);
            buf ^= 1;
        }
    }

    // Epilogue. ISA 16x16 f32 C/D layout: VGPR r -> row r (lanes 0-15) /
    // row r+8 (lanes 16-31); column = lane & 15.
    #pragma unroll
    for (int i = 0; i < 4; ++i) {
        #pragma unroll
        for (int j = 0; j < 4; ++j) {
            const int col = blockN + wx * 64 + j * 16 + (lane & 15);
            const float bv = bias ? bias[col] : 0.0f;
            const int rbase = blockM + wy * 64 + i * 16 + ((lane < 16) ? 0 : 8);
            #pragma unroll
            for (int r = 0; r < 8; ++r) {
                float v = acc[i][j][r] + bv;
                if (ACT == 1) v = fmaxf(v, 0.0f);
                C[(size_t)(rbase + r) * ldc + col] = (CT)v;
            }
        }
    }
}

// ---------------------------------------------------------------------------
// Host-side orchestration
// ---------------------------------------------------------------------------
extern "C" void kernel_launch(void* const* d_in, const int* in_sizes, int n_in,
                              void* d_out, int out_size, void* d_ws, size_t ws_size,
                              hipStream_t stream) {
    (void)in_sizes; (void)n_in; (void)out_size; (void)ws_size;
    constexpr int Bb = 4, S = 2048, D = 1024, H = 4096;
    constexpr int M = Bb * S; // 8192 flattened token rows

    const float* x  = (const float*)d_in[0];
    const float* Wq = (const float*)d_in[1];
    const float* bq = (const float*)d_in[2];
    const float* Wk = (const float*)d_in[3];
    const float* bk = (const float*)d_in[4];
    const float* Wv = (const float*)d_in[5];
    const float* bv = (const float*)d_in[6];
    const float* W1 = (const float*)d_in[7];
    const float* b1 = (const float*)d_in[8];
    const float* W2 = (const float*)d_in[9];
    const float* b2 = (const float*)d_in[10];
    const float* W3 = (const float*)d_in[11];
    const float* b3 = (const float*)d_in[12];
    float* out = (float*)d_out;

    // workspace layout (~280 MB)
    char* w = (char*)d_ws;
    size_t o = 0;
    auto take = [&](size_t bytes) -> char* {
        char* p = w + o;
        o += (bytes + 255) & ~(size_t)255;
        return p;
    };
    bf16* xb   = (bf16*)take(2ull * M * D);
    bf16* wqb  = (bf16*)take(2ull * D * D);
    bf16* wkb  = (bf16*)take(2ull * D * D);
    bf16* wvb  = (bf16*)take(2ull * D * D);
    bf16* w1b  = (bf16*)take(2ull * D * D);
    bf16* w2b  = (bf16*)take(2ull * H * D);
    bf16* w3b  = (bf16*)take(2ull * D * H);
    bf16* Qb   = (bf16*)take(2ull * M * D);
    bf16* Kb   = (bf16*)take(2ull * M * D);
    bf16* Vb   = (bf16*)take(2ull * M * D);
    bf16* attn = (bf16*)take(2ull * M * D);
    bf16* h1   = (bf16*)take(2ull * M * D);
    float* sim = (float*)take(4ull * Bb * S * S);
    bf16* P    = (bf16*)take(2ull * Bb * S * S);
    bf16* h2   = (bf16*)take(2ull * M * H);

    const dim3 blk(256);
    // 1) fp32 -> bf16 conversions
    cvt_f32_bf16<<<2048, blk, 0, stream>>>(x,  xb,  M * D);
    cvt_f32_bf16<<<1024, blk, 0, stream>>>(Wq, wqb, D * D);
    cvt_f32_bf16<<<1024, blk, 0, stream>>>(Wk, wkb, D * D);
    cvt_f32_bf16<<<1024, blk, 0, stream>>>(Wv, wvb, D * D);
    cvt_f32_bf16<<<1024, blk, 0, stream>>>(W1, w1b, D * D);
    cvt_f32_bf16<<<2048, blk, 0, stream>>>(W2, w2b, H * D);
    cvt_f32_bf16<<<2048, blk, 0, stream>>>(W3, w3b, D * H);

    // 2) Q/K/V projections: [8192,1024] = xb @ W^T + b
    gemm_wmma<true, 0, bf16><<<dim3(D / 256, M / 128, 1), blk, 0, stream>>>(
        xb, wqb, bq, Qb, M, D, D, D, D, D, 0, 0, 0);
    gemm_wmma<true, 0, bf16><<<dim3(D / 256, M / 128, 1), blk, 0, stream>>>(
        xb, wkb, bk, Kb, M, D, D, D, D, D, 0, 0, 0);
    gemm_wmma<true, 0, bf16><<<dim3(D / 256, M / 128, 1), blk, 0, stream>>>(
        xb, wvb, bv, Vb, M, D, D, D, D, D, 0, 0, 0);

    // 3) L2-normalize Q and K rows
    l2norm_rows<<<M, blk, 0, stream>>>(Qb, D);
    l2norm_rows<<<M, blk, 0, stream>>>(Kb, D);

    // 4) sim[b] = Qn[b] @ Kn[b]^T   (batched, f32 out)
    gemm_wmma<true, 0, float><<<dim3(S / 256, S / 128, Bb), blk, 0, stream>>>(
        Qb, Kb, nullptr, sim, S, S, D, D, D, S,
        (size_t)S * D, (size_t)S * D, (size_t)S * S);

    // 5) softmax rows -> P (bf16)
    softmax_rows<<<Bb * S, blk, 0, stream>>>(sim, P, S);

    // 6) attn[b] = P[b] @ V[b]      (batched, B non-transposed)
    gemm_wmma<false, 0, bf16><<<dim3(D / 256, S / 128, Bb), blk, 0, stream>>>(
        P, Vb, nullptr, attn, S, D, S, S, D, D,
        (size_t)S * S, (size_t)S * D, (size_t)S * D);

    // 7) MLP
    gemm_wmma<true, 1, bf16><<<dim3(D / 256, M / 128, 1), blk, 0, stream>>>(
        attn, w1b, b1, h1, M, D, D, D, D, D, 0, 0, 0);
    gemm_wmma<true, 1, bf16><<<dim3(H / 256, M / 128, 1), blk, 0, stream>>>(
        h1, w2b, b2, h2, M, H, D, D, D, H, 0, 0, 0);
    gemm_wmma<true, 0, float><<<dim3(D / 256, M / 128, 1), blk, 0, stream>>>(
        h2, w3b, b3, out, M, D, H, H, H, D, 0, 0, 0);
}
